// _RPN_90752658964501
// MI455X (gfx1250) — matrix operand order repaired and checked by reading, not compiled, not run on hardware
//
#include <hip/hip_runtime.h>
#include <hip/hip_bf16.h>
#include <math.h>

// ---------------------------------------------------------------------------
// CDNA5 / gfx1250 implementation of the 3D-RPN reference.
// Heavy path: implicit-GEMM 3D conv + head GEMMs via v_wmma_f32_16x16x32_f16
// (f16 inputs, f32 accumulate), wave32, 8-wave workgroups.
// K-order of the conv GEMM is tap-major (k = tap*256 + ic) so each BK=32 step
// has a single, scalar filter tap -> branch-free batched A gather.
// All B panels pre-transposed [N][K] so A and B LDS tiles are contiguous-row
// 2D tiles: staged by the Tensor Data Mover when available, vector loads else.
// Every WMMA fragment is a pair of 16B-aligned ds_load_b128.
// ---------------------------------------------------------------------------

typedef __attribute__((ext_vector_type(16))) _Float16 v16h;
typedef __attribute__((ext_vector_type(8)))  _Float16 v8h;
typedef __attribute__((ext_vector_type(8)))  float    v8f;
typedef __attribute__((ext_vector_type(4)))  unsigned int u32x4;
typedef __attribute__((ext_vector_type(8)))  int i32x8;
typedef __attribute__((ext_vector_type(4)))  int i32x4;

#define T_DIM   16
#define H_DIM   32
#define W_DIM   32
#define C_IN    256
#define C_OUT   512
#define SPATIAL (T_DIM * H_DIM * W_DIM)   // 16384 (also feat plane stride)
#define M_CONV  SPATIAL                   // 16384
#define K_CONV  (27 * C_IN)               // 6912, k = tap*256 + ic
#define N3_PAD  256                       // 54 cls + 162 bbox, padded
#define N2_PAD  64                        // 18 cls + 36 bbox, padded
#define NPROP3  (H_DIM * W_DIM * 27)      // 27648
#define NPROP2  (H_DIM * W_DIM * 9)       // 9216
#define PRE_NMS 1000
#define POST_NMS 128
#define NMS_TH  0.7f

#define SHUF16(lo, hi) __builtin_shufflevector(lo, hi, 0, 1, 2, 3, 4, 5, 6, 7, \
                                               8, 9, 10, 11, 12, 13, 14, 15)

// ---- Tensor Data Mover availability (probe-verified builtin, arity differs) --
#if defined(__has_builtin)
#if __has_builtin(__builtin_amdgcn_tensor_load_to_lds) && \
    __has_builtin(__builtin_amdgcn_s_wait_tensorcnt)
#define USE_TDM 1
#endif
#endif
#ifndef USE_TDM
#define USE_TDM 0
#warning "CDNA5 TDM builtins not available on this toolchain; using vector-load staging"
#endif

#if USE_TDM
// 2D TDM tile load: tile (tile0 x tile1) of 2-byte elements from a row-major
// tensor (row length dim0, row count dim1, row stride stride0) into LDS.
// Descriptor bit packing per CDNA5 ISA D# group0/group1.
__device__ __forceinline__ void tdm_load_2d_f16(unsigned lds_addr,
                                                unsigned long long gaddr,
                                                int dim0, int dim1,
                                                int tile0, int tile1,
                                                int stride0) {
  u32x4 g0;
  g0.x = 0x1u;                                         // count=1, user mode
  g0.y = lds_addr;                                     // bits 63:32
  g0.z = (unsigned)(gaddr & 0xffffffffull);            // global_addr[31:0]
  g0.w = (unsigned)((gaddr >> 32) & 0x01ffffffull)     // global_addr[56:32]
         | 0x80000000u;                                // type=2 (bits 127:126)
  i32x8 g1;
  g1[0] = 0x00010000;                                  // data_size=1 (2B), mask=0
  g1[1] = (dim0 & 0xffff) << 16;                       // tensor_dim0 lo16 @63:48
  g1[2] = ((dim0 >> 16) & 0xffff) | ((dim1 & 0xffff) << 16);
  g1[3] = ((dim1 >> 16) & 0xffff) | ((tile0 & 0xffff) << 16);  // tile_dim0
  g1[4] = (tile1 & 0xffff);                            // tile_dim1, tile_dim2=0
  g1[5] = stride0;                                     // tensor_dim0_stride lo32
  g1[6] = 0;
  g1[7] = 0;
  i32x4 z4 = {0, 0, 0, 0};
#if __clang_major__ >= 23
  i32x8 z8 = {0, 0, 0, 0, 0, 0, 0, 0};
  __builtin_amdgcn_tensor_load_to_lds(g0, g1, z4, z4, z8, 0);
#else
  __builtin_amdgcn_tensor_load_to_lds(g0, g1, z4, z4, 0);
#endif
}
#endif

// ---------------------------------------------------------------------------
// W_conv (512,256,3,3,3) fp32 -> BwT[oc][k] f16, k = tap*256 + ic.
// grid = (27, 512)
// ---------------------------------------------------------------------------
__global__ __launch_bounds__(256) void pack_conv_w(const float* __restrict__ W,
                                                   _Float16* __restrict__ BwT) {
  int k  = blockIdx.x * 256 + threadIdx.x;     // 0..6911
  int oc = blockIdx.y;                         // 0..511
  if (k >= K_CONV) return;
  int r = k >> 8, ic = k & 255;                // tap, input channel
  BwT[(size_t)oc * K_CONV + k] = (_Float16)W[(size_t)(oc * C_IN + ic) * 27 + r];
}

// W_cls (54,512) + W_bbox (162,512) -> BT3[n][k] f16 (zero padded) + bias
__global__ __launch_bounds__(256) void pack_head3(const float* __restrict__ Wc,
                                                  const float* __restrict__ bc,
                                                  const float* __restrict__ Wb,
                                                  const float* __restrict__ bb,
                                                  _Float16* __restrict__ BT,
                                                  float* __restrict__ bias) {
  int i = blockIdx.x * 256 + threadIdx.x;      // over 256*512
  if (i >= N3_PAD * C_OUT) return;
  int n = i >> 9, k = i & 511;
  float v = 0.f;
  if (n < 54)       v = Wc[n * C_OUT + k];
  else if (n < 216) v = Wb[(n - 54) * C_OUT + k];
  BT[(size_t)n * C_OUT + k] = (_Float16)v;
  if (k == 0) bias[n] = (n < 54) ? bc[n] : ((n < 216) ? bb[n - 54] : 0.f);
}

// W_cls16 (18,512) + W_bbox16 (36,512) -> BT2[n][k] f16 + bias
__global__ __launch_bounds__(256) void pack_head2(const float* __restrict__ Wc,
                                                  const float* __restrict__ bc,
                                                  const float* __restrict__ Wb,
                                                  const float* __restrict__ bb,
                                                  _Float16* __restrict__ BT,
                                                  float* __restrict__ bias) {
  int i = blockIdx.x * 256 + threadIdx.x;      // over 64*512
  if (i >= N2_PAD * C_OUT) return;
  int n = i >> 9, k = i & 511;
  float v = 0.f;
  if (n < 18)      v = Wc[n * C_OUT + k];
  else if (n < 54) v = Wb[(n - 18) * C_OUT + k];
  BT[(size_t)n * C_OUT + k] = (_Float16)v;
  if (k == 0) bias[n] = (n < 18) ? bc[n] : ((n < 54) ? bb[n - 18] : 0.f);
}

// ---------------------------------------------------------------------------
// 3D conv as implicit GEMM, WMMA f16 -> f32.  Block tile 128x128, BK=32.
// 8 waves: 4 (M) x 2 (N); each wave computes 32x64 = 2x4 WMMA tiles.
// As [128][32] row-major (64B rows), Bt [128][32] (rows of pre-transposed
// weight panel): fragments are 16B-aligned ds_load_b128 pairs.
// ---------------------------------------------------------------------------
__global__ __launch_bounds__(256) void conv3d_wmma(const float* __restrict__ feat,   // [256][16][32][32]
                                                   const _Float16* __restrict__ BwT, // [512][6912]
                                                   const float* __restrict__ bias,   // [512]
                                                   _Float16* __restrict__ out) {     // [16384][512]
  __shared__ _Float16 As[128][32];
  __shared__ _Float16 Bt[128][32];

  const int tid  = threadIdx.x;
  const int lane = tid & 31;
  const int wid  = tid >> 5;
  const int wm   = wid & 3;
  const int wn   = wid >> 2;
  const int blockM = blockIdx.x * 128;
  const int blockN = blockIdx.y * 128;

  // staging coordinates (fixed per thread): each thread owns half a row
  const int srow = tid >> 1;                 // 0..127
  const int skk0 = (tid & 1) * 16;           // 0 / 16
  const int am   = blockM + srow;
  const int at   = am >> 10, ah = (am >> 5) & 31, aw = am & 31;

  v8f acc[2][4];
  const v8f vzero = {0.f, 0.f, 0.f, 0.f, 0.f, 0.f, 0.f, 0.f};
  for (int i = 0; i < 2; ++i)
    for (int j = 0; j < 4; ++j) acc[i][j] = vzero;

  for (int k0 = 0; k0 < K_CONV; k0 += 32) {
#if USE_TDM
    if (wid == 0) {
      unsigned lds = (unsigned)(uintptr_t)&Bt[0][0];
      unsigned long long ga =
          (unsigned long long)(uintptr_t)(BwT + (size_t)blockN * K_CONV + k0);
      tdm_load_2d_f16(lds, ga, K_CONV, C_OUT, /*tile0=*/32, /*tile1=*/128,
                      /*stride0=*/K_CONV);
    }
#endif
    // ---- stage A half-row: single tap per K-step (k = tap*256 + ic) ----
    {
      const int r  = k0 >> 8;                        // filter tap, scalar
      const int kt = r / 9, r2 = r - kt * 9;
      const int kh = r2 / 3, kw = r2 - kh * 3;
      const int zt = at + kt - 1, zh = ah + kh - 1, zw = aw + kw - 1;
      const bool valid = ((unsigned)zt < (unsigned)T_DIM) &&
                         ((unsigned)zh < (unsigned)H_DIM) &&
                         ((unsigned)zw < (unsigned)W_DIM);
      const int ic0 = (k0 & 255) + skk0;             // input-channel base
      _Float16 tmp[16];
      if (valid) {
        const float* fp = feat + (size_t)ic0 * SPATIAL +
                          (zt * (H_DIM * W_DIM) + zh * W_DIM + zw);
#pragma unroll
        for (int i = 0; i < 16; ++i) tmp[i] = (_Float16)fp[(size_t)i * SPATIAL];
      } else {
#pragma unroll
        for (int i = 0; i < 16; ++i) tmp[i] = (_Float16)0.f;
      }
      *(v8h*)&As[srow][skk0]     = *(const v8h*)&tmp[0];
      *(v8h*)&As[srow][skk0 + 8] = *(const v8h*)&tmp[8];
    }
#if USE_TDM
    if (wid == 0) __builtin_amdgcn_s_wait_tensorcnt((short)0);
#else
    // ---- stage B half-row from pre-transposed panel (contiguous 32B) ----
    {
      const _Float16* src = BwT + (size_t)(blockN + srow) * K_CONV + k0 + skk0;
      *(v8h*)&Bt[srow][skk0]     = *(const v8h*)&src[0];
      *(v8h*)&Bt[srow][skk0 + 8] = *(const v8h*)&src[8];
    }
#endif
    __syncthreads();

    // ---- fragment loads per the gfx1250 16-bit A/B layouts ----
    v16h afrag[2], bfrag[4];
    const int kbA = (lane < 16) ? 0 : 8;   // lanes 0-15: K{0..7,16..23}; 16-31: K{8..15,24..31}
    for (int i2 = 0; i2 < 2; ++i2) {
      int row = wm * 32 + i2 * 16 + (lane & 15);
      v8h lo = *(const v8h*)&As[row][kbA];
      v8h hi = *(const v8h*)&As[row][16 + kbA];
      afrag[i2] = SHUF16(lo, hi);
    }
    const int kbB = (lane < 16) ? 0 : 8;   // lanes 0-15: K{0..7,16..23}; 16-31: K{8..15,24..31}
    for (int j2 = 0; j2 < 4; ++j2) {
      int col = wn * 64 + j2 * 16 + (lane & 15);
      v8h lo = *(const v8h*)&Bt[col][kbB];
      v8h hi = *(const v8h*)&Bt[col][16 + kbB];
      bfrag[j2] = SHUF16(lo, hi);
    }
    for (int i2 = 0; i2 < 2; ++i2)
      for (int j2 = 0; j2 < 4; ++j2)
        acc[i2][j2] = __builtin_amdgcn_wmma_f32_16x16x32_f16(
            false, afrag[i2], false, bfrag[j2], (short)0, acc[i2][j2], false, false);
    __syncthreads();
  }

  // ---- store: VGPR v holds M = v (+8 for lanes>=16), N = lane&15 ----
  const int hiOff = (lane >= 16) ? 8 : 0;
  const int ccol  = lane & 15;
  for (int i2 = 0; i2 < 2; ++i2)
    for (int j2 = 0; j2 < 4; ++j2)
      for (int v = 0; v < 8; ++v) {
        int m = blockM + wm * 32 + i2 * 16 + v + hiOff;
        int n = blockN + wn * 64 + j2 * 16 + ccol;
        float x = acc[i2][j2][v] + bias[n];
        out[(size_t)m * C_OUT + n] = (_Float16)fmaxf(x, 0.f);
      }
}

// ---------------------------------------------------------------------------
// Generic f16 GEMM + bias -> f32 (head convs).  BM=128, BN=64, BK=32.
// A [M][K] row-major, BT [N][K] pre-transposed.  Both panels staged by TDM
// (waves 0/1 issue tensor_load_to_lds, each waits TENSORcnt) when available.
// ---------------------------------------------------------------------------
__global__ __launch_bounds__(256) void gemm_f16_wmma(const _Float16* __restrict__ A,
                                                     const _Float16* __restrict__ BT,
                                                     const float* __restrict__ bias,
                                                     float* __restrict__ C,
                                                     int M, int K, int N) {
  __shared__ _Float16 As[128][32];
  __shared__ _Float16 Bt[64][32];

  const int tid  = threadIdx.x;
  const int lane = tid & 31;
  const int wid  = tid >> 5;                 // each wave owns 16 output rows
  const int blockM = blockIdx.x * 128;
  const int blockN = blockIdx.y * 64;

  v8f acc[4];
  const v8f vzero = {0.f, 0.f, 0.f, 0.f, 0.f, 0.f, 0.f, 0.f};
  for (int j = 0; j < 4; ++j) acc[j] = vzero;

  for (int k0 = 0; k0 < K; k0 += 32) {
#if USE_TDM
    if (wid == 0) {
      unsigned lds = (unsigned)(uintptr_t)&As[0][0];
      unsigned long long ga =
          (unsigned long long)(uintptr_t)(A + (size_t)blockM * K + k0);
      tdm_load_2d_f16(lds, ga, K, M, /*tile0=*/32, /*tile1=*/128, /*stride0=*/K);
    } else if (wid == 1) {
      unsigned lds = (unsigned)(uintptr_t)&Bt[0][0];
      unsigned long long ga =
          (unsigned long long)(uintptr_t)(BT + (size_t)blockN * K + k0);
      tdm_load_2d_f16(lds, ga, K, N, /*tile0=*/32, /*tile1=*/64, /*stride0=*/K);
    }
    if (wid < 2) __builtin_amdgcn_s_wait_tensorcnt((short)0);
#else
    {
      int row = tid >> 1, kk0 = (tid & 1) * 16;
      const _Float16* src = A + (size_t)(blockM + row) * K + k0 + kk0;
      *(v8h*)&As[row][kk0]     = *(const v8h*)&src[0];
      *(v8h*)&As[row][kk0 + 8] = *(const v8h*)&src[8];
    }
    {
      int col = tid >> 2, kk0 = (tid & 3) * 8;
      const _Float16* src = BT + (size_t)(blockN + col) * K + k0 + kk0;
      *(v8h*)&Bt[col][kk0] = *(const v8h*)&src[0];
    }
#endif
    __syncthreads();

    v16h afrag, bfrag[4];
    const int kbA = (lane < 16) ? 0 : 8;
    {
      int row = wid * 16 + (lane & 15);
      v8h lo = *(const v8h*)&As[row][kbA];
      v8h hi = *(const v8h*)&As[row][16 + kbA];
      afrag = SHUF16(lo, hi);
    }
    const int kbB = (lane < 16) ? 0 : 8;
    for (int j2 = 0; j2 < 4; ++j2) {
      int col = j2 * 16 + (lane & 15);
      v8h lo = *(const v8h*)&Bt[col][kbB];
      v8h hi = *(const v8h*)&Bt[col][16 + kbB];
      bfrag[j2] = SHUF16(lo, hi);
    }
    for (int j2 = 0; j2 < 4; ++j2)
      acc[j2] = __builtin_amdgcn_wmma_f32_16x16x32_f16(
          false, afrag, false, bfrag[j2], (short)0, acc[j2], false, false);
    __syncthreads();
  }

  const int hiOff = (lane >= 16) ? 8 : 0;
  const int ccol  = lane & 15;
  for (int j2 = 0; j2 < 4; ++j2)
    for (int v = 0; v < 8; ++v) {
      int m = blockM + wid * 16 + v + hiOff;
      int n = blockN + j2 * 16 + ccol;
      C[(size_t)m * N + n] = acc[j2][v] + bias[n];
    }
}

// ---------------------------------------------------------------------------
// feat2 = mean over T of conv1 (f16 in, f16 out) -> [1024][512]
// ---------------------------------------------------------------------------
__global__ __launch_bounds__(256) void mean_t_kernel(const _Float16* __restrict__ conv1,
                                                     _Float16* __restrict__ feat2) {
  int i = blockIdx.x * 256 + threadIdx.x;     // over 1024*512
  if (i >= 1024 * C_OUT) return;
  int hw = i >> 9, c = i & 511;
  float s = 0.f;
  for (int t = 0; t < T_DIM; ++t)
    s += (float)conv1[(size_t)((t << 10) + hw) * C_OUT + c];
  feat2[i] = (_Float16)(s * (1.f / 16.f));
}

// ---------------------------------------------------------------------------
// 3D path: softmax-fg mean over T, delta mean over T, anchor decode + clip.
// ---------------------------------------------------------------------------
__global__ __launch_bounds__(256) void assemble3(const float* __restrict__ out3,   // [16384][256]
                                                 const float* __restrict__ im_info,
                                                 float* __restrict__ sc,
                                                 float* __restrict__ sc_mut,
                                                 float* __restrict__ boxes) {      // [NPROP3][6]
  int n = blockIdx.x * 256 + threadIdx.x;
  if (n >= NPROP3) return;
  int hw = n / 27, a = n - hw * 27;
  int h = hw >> 5, w = hw & 31;

  float fg = 0.f, dl[6] = {0.f, 0.f, 0.f, 0.f, 0.f, 0.f};
  for (int t = 0; t < T_DIM; ++t) {
    const float* row = out3 + (size_t)((t << 10) + hw) * N3_PAD;
    float x0 = row[a], x1 = row[27 + a];
    float mx = fmaxf(x0, x1);
    float e0 = __expf(x0 - mx), e1 = __expf(x1 - mx);
    fg += e1 / (e0 + e1);
    for (int d = 0; d < 6; ++d) dl[d] += row[54 + a * 6 + d];
  }
  fg *= (1.f / 16.f);
  for (int d = 0; d < 6; ++d) dl[d] *= (1.f / 16.f);

  const float ratios[3] = {0.5f, 1.f, 2.f};
  const float scales[3] = {4.f, 8.f, 16.f};
  const float depths[3] = {16.f, 8.f, 4.f};
  int id = a / 9, aj = a - id * 9, ir = aj / 3, is = aj - ir * 3;
  float rr = ratios[ir], ss = scales[is];
  float wsz = rintf(sqrtf(256.f / rr));
  float hsz = rintf(wsz * rr);
  float aw = wsz * ss, ah = hsz * ss;
  float sx = (float)w * 16.f, sy = (float)h * 16.f;
  float ax1 = 7.5f - 0.5f * (aw - 1.f) + sx, ax2 = 7.5f + 0.5f * (aw - 1.f) + sx;
  float ay1 = 7.5f - 0.5f * (ah - 1.f) + sy, ay2 = 7.5f + 0.5f * (ah - 1.f) + sy;
  float az1 = 0.f, az2 = depths[id] - 1.f;

  float szx = ax2 - ax1 + 1.f, szy = ay2 - ay1 + 1.f, szz = az2 - az1 + 1.f;
  float cx = ax1 + 0.5f * (szx - 1.f), cy = ay1 + 0.5f * (szy - 1.f), cz = az1 + 0.5f * (szz - 1.f);
  float pcx = dl[0] * szx + cx, pcy = dl[1] * szy + cy, pcz = dl[2] * szz + cz;
  float psx = __expf(dl[3]) * szx, psy = __expf(dl[4]) * szy, psz = __expf(dl[5]) * szz;

  float imh = im_info[0] - 1.f, imw = im_info[1] - 1.f, zmx = 15.f;
  float b0 = fminf(fmaxf(pcx - 0.5f * (psx - 1.f), 0.f), imw);
  float b1 = fminf(fmaxf(pcy - 0.5f * (psy - 1.f), 0.f), imh);
  float b2 = fminf(fmaxf(pcz - 0.5f * (psz - 1.f), 0.f), zmx);
  float b3 = fminf(fmaxf(pcx + 0.5f * (psx - 1.f), 0.f), imw);
  float b4 = fminf(fmaxf(pcy + 0.5f * (psy - 1.f), 0.f), imh);
  float b5 = fminf(fmaxf(pcz + 0.5f * (psz - 1.f), 0.f), zmx);
  float* bp = boxes + (size_t)n * 6;
  bp[0] = b0; bp[1] = b1; bp[2] = b2; bp[3] = b3; bp[4] = b4; bp[5] = b5;
  sc[n] = fg; sc_mut[n] = fg;
}

// 2D path
__global__ __launch_bounds__(256) void assemble2(const float* __restrict__ out2,   // [1024][64]
                                                 const float* __restrict__ im_info,
                                                 float* __restrict__ sc,
                                                 float* __restrict__ sc_mut,
                                                 float* __restrict__ boxes) {      // [NPROP2][4]
  int n = blockIdx.x * 256 + threadIdx.x;
  if (n >= NPROP2) return;
  int hw = n / 9, a = n - hw * 9;
  int h = hw >> 5, w = hw & 31;

  const float* row = out2 + (size_t)hw * N2_PAD;
  float x0 = row[a], x1 = row[9 + a];
  float mx = fmaxf(x0, x1);
  float e0 = __expf(x0 - mx), e1 = __expf(x1 - mx);
  float fg = e1 / (e0 + e1);
  float dl[4];
  for (int d = 0; d < 4; ++d) dl[d] = row[18 + a * 4 + d];

  const float ratios[3] = {0.5f, 1.f, 2.f};
  const float scales[3] = {4.f, 8.f, 16.f};
  int ir = a / 3, is = a - ir * 3;
  float rr = ratios[ir], ss = scales[is];
  float wsz = rintf(sqrtf(256.f / rr));
  float hsz = rintf(wsz * rr);
  float aw = wsz * ss, ah = hsz * ss;
  float sx = (float)w * 16.f, sy = (float)h * 16.f;
  float ax1 = 7.5f - 0.5f * (aw - 1.f) + sx, ax2 = 7.5f + 0.5f * (aw - 1.f) + sx;
  float ay1 = 7.5f - 0.5f * (ah - 1.f) + sy, ay2 = 7.5f + 0.5f * (ah - 1.f) + sy;

  float szx = ax2 - ax1 + 1.f, szy = ay2 - ay1 + 1.f;
  float cx = ax1 + 0.5f * (szx - 1.f), cy = ay1 + 0.5f * (szy - 1.f);
  float pcx = dl[0] * szx + cx, pcy = dl[1] * szy + cy;
  float psx = __expf(dl[2]) * szx, psy = __expf(dl[3]) * szy;

  float imh = im_info[0] - 1.f, imw = im_info[1] - 1.f;
  float b0 = fminf(fmaxf(pcx - 0.5f * (psx - 1.f), 0.f), imw);
  float b1 = fminf(fmaxf(pcy - 0.5f * (psy - 1.f), 0.f), imh);
  float b2 = fminf(fmaxf(pcx + 0.5f * (psx - 1.f), 0.f), imw);
  float b3 = fminf(fmaxf(pcy + 0.5f * (psy - 1.f), 0.f), imh);
  float* bp = boxes + (size_t)n * 4;
  bp[0] = b0; bp[1] = b1; bp[2] = b2; bp[3] = b3;
  sc[n] = fg; sc_mut[n] = fg;
}

// ---------------------------------------------------------------------------
// Iterative top-K by block-wide argmax (single workgroup, destroys sc_mut).
// ---------------------------------------------------------------------------
__global__ __launch_bounds__(1024) void topk_kernel(float* __restrict__ sc_mut, int N,
                                                    int K, int* __restrict__ idx_out) {
  __shared__ float sv[1024];
  __shared__ int   si[1024];
  const int tid = threadIdx.x;
  for (int k = 0; k < K; ++k) {
    float best = -3.4e38f; int bi = 0;
    for (int j = tid; j < N; j += 1024) {
      float v = sc_mut[j];
      if (v > best) { best = v; bi = j; }
    }
    sv[tid] = best; si[tid] = bi;
    __syncthreads();
    for (int s = 512; s > 0; s >>= 1) {
      if (tid < s && sv[tid + s] > sv[tid]) { sv[tid] = sv[tid + s]; si[tid] = si[tid + s]; }
      __syncthreads();
    }
    if (tid == 0) { idx_out[k] = si[0]; sc_mut[si[0]] = -3.0e38f; }
    __syncthreads();
  }
}

// ---------------------------------------------------------------------------
// Greedy NMS, mirroring reference semantics (keep=0 when exhausted).
// ---------------------------------------------------------------------------
template <int ND>
__global__ __launch_bounds__(1024) void nms_kernel(const float* __restrict__ boxes,
                                                   const int* __restrict__ topk,
                                                   int* __restrict__ keep) {
  __shared__ unsigned char supp[PRE_NMS];
  __shared__ int smin[1024];
  const int tid = threadIdx.x;
  if (tid < PRE_NMS) supp[tid] = 0;
  float myb[2 * ND];
  for (int c = 0; c < 2 * ND; ++c) myb[c] = 0.f;
  if (tid < PRE_NMS) {
    const float* p = boxes + (size_t)topk[tid] * (2 * ND);
    for (int c = 0; c < 2 * ND; ++c) myb[c] = p[c];
  }
  __syncthreads();

  for (int i = 0; i < POST_NMS; ++i) {
    smin[tid] = (tid < PRE_NMS && !supp[tid]) ? tid : 0x7fffffff;
    __syncthreads();
    for (int s = 512; s > 0; s >>= 1) {
      if (tid < s && smin[tid + s] < smin[tid]) smin[tid] = smin[tid + s];
      __syncthreads();
    }
    int first = smin[0];
    if (first == 0x7fffffff) first = 0;
    if (tid == 0) keep[i] = first;
    const float* pb = boxes + (size_t)topk[first] * (2 * ND);
    if (tid < PRE_NMS) {
      float inter = 1.f, va = 1.f, vb = 1.f;
      for (int c = 0; c < ND; ++c) {
        float lo = fmaxf(pb[c], myb[c]);
        float hi = fminf(pb[ND + c], myb[ND + c]);
        inter *= fmaxf(hi - lo + 1.f, 0.f);
        va *= pb[ND + c] - pb[c] + 1.f;
        vb *= myb[ND + c] - myb[c] + 1.f;
      }
      float iou = inter / (va + vb - inter);
      if (iou > NMS_TH) supp[tid] = 1;
      if (tid == first) supp[tid] = 1;
    }
    __syncthreads();
  }
}

// ---------------------------------------------------------------------------
// Final gather: rois (128x7) || rois16 (128x5) || 4 zero scalars.
// ---------------------------------------------------------------------------
__global__ __launch_bounds__(256) void gather_out(const float* __restrict__ boxes3,
                                                  const int* __restrict__ topk3,
                                                  const int* __restrict__ keep3,
                                                  const float* __restrict__ boxes2,
                                                  const int* __restrict__ topk2,
                                                  const int* __restrict__ keep2,
                                                  float* __restrict__ out) {
  int i = blockIdx.x * 256 + threadIdx.x;
  if (i < POST_NMS) {
    int bi = topk3[keep3[i]];
    float* o = out + (size_t)i * 7;
    o[0] = 0.f;
    for (int c = 0; c < 6; ++c) o[1 + c] = boxes3[(size_t)bi * 6 + c];
  } else if (i < 2 * POST_NMS) {
    int j = i - POST_NMS;
    int bi = topk2[keep2[j]];
    float* o = out + POST_NMS * 7 + (size_t)j * 5;
    o[0] = 0.f;
    for (int c = 0; c < 4; ++c) o[1 + c] = boxes2[(size_t)bi * 4 + c];
  } else if (i < 2 * POST_NMS + 4) {
    out[POST_NMS * 7 + POST_NMS * 5 + (i - 2 * POST_NMS)] = 0.f;
  }
}

// ---------------------------------------------------------------------------
extern "C" void kernel_launch(void* const* d_in, const int* in_sizes, int n_in,
                              void* d_out, int out_size, void* d_ws, size_t ws_size,
                              hipStream_t stream) {
  (void)in_sizes; (void)n_in; (void)out_size; (void)ws_size;
  const float* base_feat = (const float*)d_in[0];
  const float* im_info   = (const float*)d_in[1];
  // d_in[2] gt_boxes, d_in[3] gt_rois: unused by the returned outputs
  const float* W_conv  = (const float*)d_in[4];
  const float* b_conv  = (const float*)d_in[5];
  const float* W_cls   = (const float*)d_in[6];
  const float* b_cls   = (const float*)d_in[7];
  const float* W_bbox  = (const float*)d_in[8];
  const float* b_bbox  = (const float*)d_in[9];
  const float* W_cls16 = (const float*)d_in[10];
  const float* b_cls16 = (const float*)d_in[11];
  const float* W_bbox16= (const float*)d_in[12];
  const float* b_bbox16= (const float*)d_in[13];
  float* out = (float*)d_out;

  // ---- workspace carve-out (256B aligned) ----
  char* ws = (char*)d_ws;
  size_t cur = 0;
  auto alloc = [&](size_t bytes) -> void* {
    void* p = ws + cur;
    cur = (cur + bytes + 255) & ~(size_t)255;
    return p;
  };
  _Float16* BconvT = (_Float16*)alloc((size_t)C_OUT * K_CONV * 2);      // 7.1 MB
  _Float16* conv1  = (_Float16*)alloc((size_t)M_CONV * C_OUT * 2);      // 16 MB
  _Float16* BT3    = (_Float16*)alloc((size_t)N3_PAD * C_OUT * 2);
  float*    bias3  = (float*)   alloc((size_t)N3_PAD * 4);
  float*    out3   = (float*)   alloc((size_t)M_CONV * N3_PAD * 4);     // 16.8 MB
  _Float16* feat2  = (_Float16*)alloc((size_t)1024 * C_OUT * 2);
  _Float16* BT2    = (_Float16*)alloc((size_t)N2_PAD * C_OUT * 2);
  float*    bias2  = (float*)   alloc((size_t)N2_PAD * 4);
  float*    out2   = (float*)   alloc((size_t)1024 * N2_PAD * 4);
  float*    sc3    = (float*)   alloc((size_t)NPROP3 * 4);
  float*    sc3m   = (float*)   alloc((size_t)NPROP3 * 4);
  float*    boxes3 = (float*)   alloc((size_t)NPROP3 * 6 * 4);
  float*    sc2    = (float*)   alloc((size_t)NPROP2 * 4);
  float*    sc2m   = (float*)   alloc((size_t)NPROP2 * 4);
  float*    boxes2 = (float*)   alloc((size_t)NPROP2 * 4 * 4);
  int*      idx3   = (int*)     alloc((size_t)PRE_NMS * 4);
  int*      keep3  = (int*)     alloc((size_t)POST_NMS * 4);
  int*      idx2   = (int*)     alloc((size_t)PRE_NMS * 4);
  int*      keep2  = (int*)     alloc((size_t)POST_NMS * 4);

  // ---- 1. pack weights to transposed f16 GEMM panels ----
  pack_conv_w<<<dim3(K_CONV / 256, C_OUT), 256, 0, stream>>>(W_conv, BconvT);
  pack_head3 <<<(N3_PAD * C_OUT + 255) / 256, 256, 0, stream>>>(W_cls, b_cls, W_bbox, b_bbox, BT3, bias3);
  pack_head2 <<<(N2_PAD * C_OUT + 255) / 256, 256, 0, stream>>>(W_cls16, b_cls16, W_bbox16, b_bbox16, BT2, bias2);

  // ---- 2. 3D conv (implicit GEMM, WMMA) -> relu f16 activation ----
  conv3d_wmma<<<dim3(M_CONV / 128, C_OUT / 128), 256, 0, stream>>>(base_feat, BconvT, b_conv, conv1);

  // ---- 3. cls3/bbox3 head GEMM: [16384x512] x [512x256] ----
  gemm_f16_wmma<<<dim3(M_CONV / 128, N3_PAD / 64), 256, 0, stream>>>(conv1, BT3, bias3, out3,
                                                                     M_CONV, C_OUT, N3_PAD);
  // ---- 4. T-mean + cls2/bbox2 head GEMM: [1024x512] x [512x64] ----
  mean_t_kernel<<<(1024 * C_OUT + 255) / 256, 256, 0, stream>>>(conv1, feat2);
  gemm_f16_wmma<<<dim3(1024 / 128, N2_PAD / 64), 256, 0, stream>>>(feat2, BT2, bias2, out2,
                                                                   1024, C_OUT, N2_PAD);

  // ---- 5. softmax / means / decode / clip ----
  assemble3<<<(NPROP3 + 255) / 256, 256, 0, stream>>>(out3, im_info, sc3, sc3m, boxes3);
  assemble2<<<(NPROP2 + 255) / 256, 256, 0, stream>>>(out2, im_info, sc2, sc2m, boxes2);

  // ---- 6. top-1000 + greedy NMS(128) per path ----
  topk_kernel<<<1, 1024, 0, stream>>>(sc3m, NPROP3, PRE_NMS, idx3);
  topk_kernel<<<1, 1024, 0, stream>>>(sc2m, NPROP2, PRE_NMS, idx2);
  nms_kernel<3><<<1, 1024, 0, stream>>>(boxes3, idx3, keep3);
  nms_kernel<2><<<1, 1024, 0, stream>>>(boxes2, idx2, keep2);

  // ---- 7. gather rois | rois16 | 4 zeros ----
  gather_out<<<2, 256, 0, stream>>>(boxes3, idx3, keep3, boxes2, idx2, keep2, out);
}
